// MoE_25409026523805
// MI455X (gfx1250) — matrix operand order repaired and checked by reading, not compiled, not run on hardware
//
#include <hip/hip_runtime.h>
#include <hip/hip_bf16.h>
#include <stdint.h>

typedef __attribute__((ext_vector_type(16))) __bf16 v16bf;
typedef __attribute__((ext_vector_type(8)))  float  v8f;
typedef __attribute__((ext_vector_type(4)))  __bf16 v4bf;

#define T_TOK 8192
#define DIM   2048
#define EDIM  8192
#define NEXP  8

#define BM 128
#define BN 128
#define BK 64
#define LDSS 72   // padded LDS row stride (bf16 elems) to spread banks

#if defined(__gfx1250__) && __has_builtin(__builtin_amdgcn_global_load_async_to_lds_b128) && __has_builtin(__builtin_amdgcn_s_wait_asynccnt)
#define MOE_ASYNC_LDS 1
#endif

union Frag {
  v16bf v;
  uint4 u[2];
};

// Exact parameter types for the async-to-LDS builtin (from hipcc diagnostic:
// param 1 is 'int __vector(4) __device__ *', i.e. int4 in AS(1); LDS side AS(3)).
typedef int moe_v4i __attribute__((vector_size(16)));
typedef __attribute__((address_space(1))) moe_v4i* glb_v4i_ptr;
typedef __attribute__((address_space(3))) moe_v4i* lds_v4i_ptr;

// 16-byte global -> LDS copy via the CDNA5 async-to-LDS unit (ASYNCcnt).
// LDS flat addresses carry the offset in the low 32 bits -> truncate for AS(3).
__device__ __forceinline__ void copy16_g2s(__bf16* dst_lds, const __bf16* src_g) {
#ifdef MOE_ASYNC_LDS
  __builtin_amdgcn_global_load_async_to_lds_b128(
      (glb_v4i_ptr)(uintptr_t)src_g,
      (lds_v4i_ptr)(uint32_t)(uintptr_t)dst_lds,
      0, 0);
#else
  *(uint4*)dst_lds = *(const uint4*)src_g;
#endif
}

// Issue one 128x64 A tile + 128x64 B tile (8 async b128 per thread => 8 per wave).
__device__ __forceinline__ void stage_tile(__bf16 (*As)[LDSS], __bf16 (*Bs)[LDSS],
                                           const __bf16* Ag, const __bf16* Bg,
                                           int K, int kb, int tid) {
#pragma unroll
  for (int i = 0; i < 4; ++i) {
    int c = tid + i * 256;
    int row = c >> 3;
    int col = (c & 7) * 8;
    copy16_g2s(&As[row][col], Ag + (size_t)row * K + kb + col);
    copy16_g2s(&Bs[row][col], Bg + (size_t)row * K + kb + col);
  }
}

// ---------------- fp32 -> bf16 ----------------
__global__ __launch_bounds__(256) void f32_to_bf16_kernel(const float* __restrict__ in,
                                                          __bf16* __restrict__ out,
                                                          long n) {
  long i = ((long)blockIdx.x * blockDim.x + threadIdx.x) * 4;
  if (i + 3 < n) {
    float4 v = *(const float4*)(in + i);
    v4bf o;
    o.x = (__bf16)v.x;
    o.y = (__bf16)v.y;
    o.z = (__bf16)v.z;
    o.w = (__bf16)v.w;
    *(v4bf*)(out + i) = o;
  }
}

// ---------------- gating: scale[t] = s/(s+1e-9), s = top-2 softmax mass ----
__global__ __launch_bounds__(256) void gate_scale_kernel(const float* __restrict__ x,
                                                         const float* __restrict__ gw,
                                                         float* __restrict__ scale) {
  const int wave = threadIdx.x >> 5;
  const int lane = threadIdx.x & 31;
  const int t = blockIdx.x * 8 + wave;
  if (t >= T_TOK) return;
  const float* xr = x + (size_t)t * DIM;

  float logit[NEXP];
#pragma unroll
  for (int e = 0; e < NEXP; ++e) {
    const float* wr = gw + (size_t)e * DIM;
    float s = 0.f;
    for (int d = lane; d < DIM; d += 32) s = __builtin_fmaf(xr[d], wr[d], s);
#pragma unroll
    for (int off = 16; off >= 1; off >>= 1) s += __shfl_xor(s, off, 32);
    logit[e] = s;  // all lanes hold the full sum
  }
  float mx = logit[0];
#pragma unroll
  for (int e = 1; e < NEXP; ++e) mx = fmaxf(mx, logit[e]);
  float se = 0.f, m1 = -3.4e38f, m2 = -3.4e38f;
#pragma unroll
  for (int e = 0; e < NEXP; ++e) {
    se += __expf(logit[e] - mx);
    if (logit[e] > m1) { m2 = m1; m1 = logit[e]; }
    else if (logit[e] > m2) { m2 = logit[e]; }
  }
  float s2 = (__expf(m1 - mx) + __expf(m2 - mx)) / se;
  if (lane == 0) scale[t] = s2 / (s2 + 1e-9f);
}

// ---------------- bf16 WMMA GEMM:  D[M][N] = A[M][K] * Bw[N][K]^T ----------
// Double-buffered async-to-LDS pipeline: tile i+1 streams in (ASYNCcnt) while
// tile i is consumed by WMMA. s_wait_asynccnt 8 drains exactly the current
// tile's 8 in-order copies, leaving the next tile's 8 in flight.
// EPILOGUE 0: SiLU, store bf16.   EPILOGUE 1: multiply by scale[row], store f32.
template <int EPILOGUE>
__global__ __launch_bounds__(256) void moe_gemm_kernel(
    const __bf16* __restrict__ A,   // [M][K] row-major
    const __bf16* __restrict__ Bw,  // [N][K] row-major (K contiguous)
    void* __restrict__ Out,
    const float* __restrict__ scale,
    int M, int N, int K) {
  __shared__ __align__(16) __bf16 As[2][BM][LDSS];
  __shared__ __align__(16) __bf16 Bs[2][BN][LDSS];

  const int tid  = threadIdx.x;
  const int wave = tid >> 5;
  const int lane = tid & 31;
  const int wm = wave >> 1;  // 0..3 : 32-row strip
  const int wn = wave & 1;   // 0..1 : 64-col strip
  const int ln = lane & 15;
  const int lh = lane >> 4;

  const int mBase = blockIdx.y * BM;
  const int nBase = blockIdx.x * BN;

  v8f acc[2][4] = {};

  const __bf16* Ag = A  + (size_t)mBase * K;
  const __bf16* Bg = Bw + (size_t)nBase * K;

  const int nk = K / BK;
  stage_tile(As[0], Bs[0], Ag, Bg, K, 0, tid);  // prologue: tile 0 in flight

  for (int ib = 0; ib < nk; ++ib) {
    const int cur = ib & 1;
    if (ib + 1 < nk) {
      // Issue next tile into the other buffer (safe: last reader of that
      // buffer passed the trailing barrier of iteration ib-1).
      stage_tile(As[cur ^ 1], Bs[cur ^ 1], Ag, Bg, K, (ib + 1) * BK, tid);
#ifdef MOE_ASYNC_LDS
      __builtin_amdgcn_s_wait_asynccnt(8);   // current tile landed, next still in flight
#endif
    } else {
#ifdef MOE_ASYNC_LDS
      __builtin_amdgcn_s_wait_asynccnt(0);   // drain final tile
#endif
    }
    __syncthreads();

    if (ib + 2 < nk) {  // warm L2 for tile i+2 (global_prefetch_b8)
      int r = tid >> 1;
      int c = (tid & 1) * 32;
      __builtin_prefetch(Ag + (size_t)r * K + (ib + 2) * BK + c, 0, 1);
      __builtin_prefetch(Bg + (size_t)r * K + (ib + 2) * BK + c, 0, 1);
    }

#pragma unroll
    for (int kk = 0; kk < BK; kk += 32) {
      Frag a[2], b[4];
      // A 16x32 bf16 layout: lanes 0-15 row M=ln hold K = lh*8..+7 (v0-3)
      // and K = 16+lh*8..+7 (v4-7).
#pragma unroll
      for (int mt = 0; mt < 2; ++mt) {
        const __bf16* p = &As[cur][wm * 32 + mt * 16 + ln][kk + lh * 8];
        a[mt].u[0] = *(const uint4*)p;
        a[mt].u[1] = *(const uint4*)(p + 16);
      }
      // B 32x16 layout: lane column N=ln holds K = lh*8..+7 and K = 16+lh*8..+7.
#pragma unroll
      for (int nt = 0; nt < 4; ++nt) {
        const __bf16* p = &Bs[cur][wn * 64 + nt * 16 + ln][kk + lh * 8];
        b[nt].u[0] = *(const uint4*)p;
        b[nt].u[1] = *(const uint4*)(p + 16);
      }
#pragma unroll
      for (int mt = 0; mt < 2; ++mt)
#pragma unroll
        for (int nt = 0; nt < 4; ++nt)
          acc[mt][nt] = __builtin_amdgcn_wmma_f32_16x16x32_bf16(
              false, a[mt].v, false, b[nt].v, (short)0, acc[mt][nt], false, false);
    }
    __syncthreads();  // all waves done with buffer `cur` before it is re-staged
  }

  // C/D layout: VGPR v, lanes 0-15 -> M=v, lanes 16-31 -> M=v+8; N=ln.
#pragma unroll
  for (int mt = 0; mt < 2; ++mt) {
#pragma unroll
    for (int nt = 0; nt < 4; ++nt) {
#pragma unroll
      for (int v = 0; v < 8; ++v) {
        int row = mBase + wm * 32 + mt * 16 + lh * 8 + v;
        int col = nBase + wn * 64 + nt * 16 + ln;
        float val = acc[mt][nt][v];
        if (EPILOGUE == 0) {
          float sv = val / (1.0f + __expf(-val));  // SiLU
          ((__bf16*)Out)[(size_t)row * N + col] = (__bf16)sv;
        } else {
          ((float*)Out)[(size_t)row * N + col] = val * scale[row];
        }
      }
    }
  }
}

extern "C" void kernel_launch(void* const* d_in, const int* in_sizes, int n_in,
                              void* d_out, int out_size, void* d_ws, size_t ws_size,
                              hipStream_t stream) {
  (void)in_sizes; (void)n_in; (void)out_size; (void)ws_size;
  const float* x      = (const float*)d_in[0];
  const float* gate_w = (const float*)d_in[1];
  const float* up_w   = (const float*)d_in[2];
  const float* down_w = (const float*)d_in[3];
  // d_in[4] is k (==2); top-2 is hardwired in the gate kernel.

  char* ws = (char*)d_ws;
  size_t off = 0;
  __bf16* xb    = (__bf16*)(ws + off); off += (size_t)T_TOK * DIM * 2;
  __bf16* upb   = (__bf16*)(ws + off); off += (size_t)EDIM * DIM * 2;
  __bf16* downb = (__bf16*)(ws + off); off += (size_t)DIM * EDIM * 2;
  __bf16* Hb    = (__bf16*)(ws + off); off += (size_t)T_TOK * EDIM * 2;
  float*  scl   = (float*)(ws + off);  off += (size_t)T_TOK * 4;

  const long nx = (long)T_TOK * DIM;
  const long nu = (long)EDIM * DIM;
  const long nd = (long)DIM * EDIM;
  f32_to_bf16_kernel<<<(int)(nx / 4 / 256), 256, 0, stream>>>(x, xb, nx);
  f32_to_bf16_kernel<<<(int)(nu / 4 / 256), 256, 0, stream>>>(up_w, upb, nu);
  f32_to_bf16_kernel<<<(int)(nd / 4 / 256), 256, 0, stream>>>(down_w, downb, nd);

  gate_scale_kernel<<<T_TOK / 8, 256, 0, stream>>>(x, gate_w, scl);

  // H = silu(X * Up^T): M=8192, N=8192, K=2048
  moe_gemm_kernel<0><<<dim3(EDIM / BN, T_TOK / BM), 256, 0, stream>>>(
      xb, upb, (void*)Hb, nullptr, T_TOK, EDIM, DIM);
  // out = (H * Down^T) * scale: M=8192, N=2048, K=8192
  moe_gemm_kernel<1><<<dim3(DIM / BN, T_TOK / BM), 256, 0, stream>>>(
      Hb, downb, d_out, scl, T_TOK, DIM, EDIM);
}
